// CrossAttention_16638703305085
// MI455X (gfx1250) — matrix operand order repaired and checked by reading, not compiled, not run on hardware
//
#include <hip/hip_runtime.h>
#include <hip/hip_bf16.h>
#include <stdint.h>

// ---------------------------------------------------------------------------
// Problem constants (fixed by the reference)
// ---------------------------------------------------------------------------
#define BB   2
#define NN_  1024
#define MM_  2048
#define DIM_ 1024
#define HH   16
#define DD   64
#define SCALE_F 0.125f   // 64^-0.5

typedef __attribute__((ext_vector_type(16))) _Float16 v16h;
typedef __attribute__((ext_vector_type(8)))  _Float16 h8;
typedef __attribute__((ext_vector_type(8)))  float    v8f;

union H8U { h8 h; uint32_t u[4]; };

__device__ __forceinline__ v16h mk16(h8 lo, h8 hi) {
    v16h r;
#pragma unroll
    for (int j = 0; j < 8; ++j) { r[j] = lo[j]; r[8 + j] = hi[j]; }
    return r;
}

__device__ __forceinline__ v8f wmma_f16(v16h a, v16h b, v8f c) {
    return __builtin_amdgcn_wmma_f32_16x16x32_f16(false, a, false, b, (short)0, c,
                                                  false, false);
}

// ---- CDNA5 async global->LDS copy (ASYNCcnt path), 16 bytes per lane -------
__device__ __forceinline__ void async_copy16(uint32_t lds_off,
                                             const _Float16* gptr) {
    asm volatile("global_load_async_to_lds_b128 %0, %1, off"
                 :
                 : "v"(lds_off), "v"(gptr)
                 : "memory");
}

template <int N>
__device__ __forceinline__ void wait_async() {
    asm volatile("s_wait_asynccnt %0" : : "i"(N) : "memory");
}

// ---------------------------------------------------------------------------
// Elementwise converts / transposes
// ---------------------------------------------------------------------------
__global__ void cvt_f32_to_f16(const float* __restrict__ s, _Float16* __restrict__ d,
                               int n) {
    int i = blockIdx.x * blockDim.x + threadIdx.x;
    if (i < n) d[i] = (_Float16)s[i];
}

// wt[o][i] = w[i][o]  (w: [I][O] fp32 row-major -> wt: [O][I] f16 row-major)
__global__ void transpose_cvt(const float* __restrict__ w, _Float16* __restrict__ wt,
                              int I, int O) {
    int idx = blockIdx.x * blockDim.x + threadIdx.x;
    if (idx < I * O) {
        int o = idx / I, i = idx % I;
        wt[idx] = (_Float16)w[(size_t)i * O + o];
    }
}

// vT[b][h][d][m] = kvraw[b][m][1024 + h*64 + d]
__global__ void v_transpose(const _Float16* __restrict__ kvraw,
                            _Float16* __restrict__ vT) {
    int idx = blockIdx.x * blockDim.x + threadIdx.x;
    if (idx >= BB * HH * DD * MM_) return;
    int m = idx % MM_;
    int rest = idx / MM_;
    int d = rest % DD;
    int bh = rest / DD;
    int h = bh % HH, b = bh / HH;
    vT[idx] = kvraw[((size_t)(b * MM_ + m)) * (2 * DIM_) + DIM_ + h * DD + d];
}

// ---------------------------------------------------------------------------
// WMMA GEMM with async-LDS staging: C[MM][NN] = A[MM][KK] * B (B given
// transposed: BT[NN][KK]).  Block = 8 waves (4x2), block tile 128x64,
// K-step 32.  Per step the block stages A[128x32] (8KB) + BT-tile[64x32]
// (4KB) into LDS via GLOBAL_LOAD_ASYNC_TO_LDS_B128 (3 x 16B per thread),
// double-buffered; fragments come from LDS (ds_load_b128).
// ---------------------------------------------------------------------------
template <typename OT>
__global__ __launch_bounds__(256) void gemm_wmma_lds(
    const _Float16* __restrict__ A, const _Float16* __restrict__ BT,
    OT* __restrict__ C, int MM, int NN, int KK) {
    // 2 buffers x (A 4096 halves + B 2048 halves) = 24 KB
    __shared__ alignas(128) _Float16 lds[2 * 6144];

    int tid = threadIdx.x;
    int w = tid >> 5, lane = tid & 31;
    int li = lane & 15, hi = lane >> 4;
    int tiles_x = NN >> 6;
    int bx = blockIdx.x % tiles_x, by = blockIdx.x / tiles_x;
    int rowB = by << 7, colB = bx << 6;
    int wy = w >> 1, wx = w & 1;

    uint32_t lbase = (uint32_t)(uintptr_t)(&lds[0]);

    auto stage = [&](int buf, int kk) {
        uint32_t base = lbase + (uint32_t)buf * 12288u;
        // A tile: 128 rows x 32 halves = 512 x 16B chunks, 2 per thread
        int c0 = tid, c1 = tid + 256;
        async_copy16(base + (uint32_t)c0 * 16u,
                     A + (size_t)(rowB + (c0 >> 2)) * KK + kk + (c0 & 3) * 8);
        async_copy16(base + (uint32_t)c1 * 16u,
                     A + (size_t)(rowB + (c1 >> 2)) * KK + kk + (c1 & 3) * 8);
        // B tile: 64 rows x 32 halves = 256 x 16B chunks, 1 per thread
        async_copy16(base + 8192u + (uint32_t)tid * 16u,
                     BT + (size_t)(colB + (tid >> 2)) * KK + kk + (tid & 3) * 8);
    };

    v8f acc00 = {}, acc01 = {}, acc10 = {}, acc11 = {};

    stage(0, 0);

    for (int kk = 0; kk < KK; kk += 32) {
        int cur = (kk >> 5) & 1;
        if (kk + 32 < KK) {
            stage(cur ^ 1, kk + 32);
            wait_async<3>();   // current stage's 3 async ops (per wave) done
        } else {
            wait_async<0>();
        }
        __syncthreads();

        const _Float16* Al = lds + cur * 6144;
        const _Float16* Bl = Al + 4096;
        const _Float16* ar0 = Al + (size_t)(wy * 32 + li) * 32;
        const _Float16* ar1 = ar0 + 16 * 32;
        const _Float16* bc0 = Bl + (size_t)(wx * 32 + li) * 32;
        const _Float16* bc1 = bc0 + 16 * 32;

        v16h a0 = mk16(*(const h8*)(ar0 + hi * 8), *(const h8*)(ar0 + 16 + hi * 8));
        v16h a1 = mk16(*(const h8*)(ar1 + hi * 8), *(const h8*)(ar1 + 16 + hi * 8));
        v16h b0 = mk16(*(const h8*)(bc0 + hi * 8), *(const h8*)(bc0 + 16 + hi * 8));
        v16h b1 = mk16(*(const h8*)(bc1 + hi * 8), *(const h8*)(bc1 + 16 + hi * 8));

        acc00 = wmma_f16(a0, b0, acc00);
        acc01 = wmma_f16(a0, b1, acc01);
        acc10 = wmma_f16(a1, b0, acc10);
        acc11 = wmma_f16(a1, b1, acc11);

        __syncthreads();       // all reads of buf `cur` done before overwrite
    }

    int row0 = rowB + wy * 32, col0 = colB + wx * 32;
#pragma unroll
    for (int r = 0; r < 8; ++r) {
        int rA = row0 + r + hi * 8;
        int rB = rA + 16;
        int c0 = col0 + li, c1 = col0 + 16 + li;
        C[(size_t)rA * NN + c0] = (OT)acc00[r];
        C[(size_t)rA * NN + c1] = (OT)acc01[r];
        C[(size_t)rB * NN + c0] = (OT)acc10[r];
        C[(size_t)rB * NN + c1] = (OT)acc11[r];
    }
}

// ---------------------------------------------------------------------------
// Per-head LayerNorm over last dim (64), one wave per (b,h,s) row.
// ---------------------------------------------------------------------------
__global__ void ln_head(const _Float16* __restrict__ raw, int S, int rowStride,
                        const float* __restrict__ w, _Float16* __restrict__ dst) {
    int wid = blockIdx.x * (blockDim.x >> 5) + (threadIdx.x >> 5);
    int s = wid % S;
    int bh = wid / S;
    int h = bh % HH, b = bh / HH;
    int lane = threadIdx.x & 31;

    const _Float16* p =
        raw + (size_t)(b * S + s) * rowStride + h * DD + lane * 2;
    float x0 = (float)p[0], x1 = (float)p[1];

    float sum = x0 + x1;
#pragma unroll
    for (int off = 16; off; off >>= 1) sum += __shfl_xor(sum, off, 32);
    float mu = sum * (1.0f / 64.0f);
    float d0 = x0 - mu, d1 = x1 - mu;
    float vs = d0 * d0 + d1 * d1;
#pragma unroll
    for (int off = 16; off; off >>= 1) vs += __shfl_xor(vs, off, 32);
    float inv = rsqrtf(vs * (1.0f / 64.0f) + 1e-6f);

    float w0 = w[h * DD + lane * 2], w1 = w[h * DD + lane * 2 + 1];
    _Float16* o = dst + (((size_t)(b * HH + h) * S + s) * DD) + lane * 2;
    o[0] = (_Float16)(d0 * inv * w0);
    o[1] = (_Float16)(d1 * inv * w1);
}

// ---------------------------------------------------------------------------
// Flash attention with async-LDS K/V staging.
// Block = 8 waves = one (b,h) and 128 consecutive n rows (8 x 16-row tiles).
// Per 32-wide m step the block stages K[32][64] and Vt[64][32] (4KB each) into
// LDS via GLOBAL_LOAD_ASYNC_TO_LDS_B128 (one 16B op per thread per tile),
// double-buffered; waves consume fragments from LDS (ds_load_b128).
// Computes S^T = K*Q^T (softmax over rows of S^T = in-lane + shfl_xor(16)),
// then O^T = V^T * P^T.
// ---------------------------------------------------------------------------
__device__ __forceinline__ float msel(uint32_t w, int r, float t) {
    return ((w >> (8 * r)) & 0xffu) ? t : -1e30f;
}

__global__ __launch_bounds__(256) void flash_attn(
    const _Float16* __restrict__ qh, const _Float16* __restrict__ kh,
    const _Float16* __restrict__ vT, const unsigned char* __restrict__ mask,
    _Float16* __restrict__ attnh) {
    // LDS: 2 buffers x (K tile 2048 halves + V tile 2048 halves) = 16 KB
    __shared__ alignas(128) _Float16 lds[2 * 4096];

    int tid = threadIdx.x;
    int wv = tid >> 5;
    int lane = tid & 31;
    int li = lane & 15, hi = lane >> 4;

    int bh = blockIdx.x >> 3;                 // 0..31
    int h = bh % HH, b = bh / HH;
    int ntile = (blockIdx.x & 7) * 8 + wv;    // 0..63
    int n = ntile * 16 + li;

    const _Float16* kbase = kh + (size_t)(b * HH + h) * MM_ * DD;
    const _Float16* vbase = vT + (size_t)(b * HH + h) * DD * MM_;
    const unsigned char* mbase = mask + ((size_t)b * NN_ + n) * MM_;

    // Q^T B-fragments (K = d, N = n): d = 8*hi+s, 16+8*hi+(s-8) per lane, loaded once.
    const _Float16* qb = qh + ((size_t)(b * HH + h) * NN_ + n) * DD;
    v16h bq0 = mk16(*(const h8*)(qb + hi * 8), *(const h8*)(qb + 16 + hi * 8));
    v16h bq1 = mk16(*(const h8*)(qb + 32 + hi * 8), *(const h8*)(qb + 48 + hi * 8));

    uint32_t lds_base = (uint32_t)(uintptr_t)(&lds[0]);

    // Stage K/V tiles for chunk m0 into buffer `buf` (all 256 threads).
    auto stage = [&](int buf, int m0) {
        uint32_t base = lds_base + (uint32_t)buf * 8192u;
        // K tile: 32 rows x 64 halves, fully linear: chunk tid -> 16 bytes
        async_copy16(base + (uint32_t)tid * 16u, kbase + (size_t)m0 * DD + tid * 8);
        // V tile: 64 rows x 32 halves; lds offset also linear in tid
        async_copy16(base + 4096u + (uint32_t)tid * 16u,
                     vbase + (size_t)(tid >> 2) * MM_ + m0 + (tid & 3) * 8);
    };

    v8f acc0 = {}, acc1 = {}, acc2 = {}, acc3 = {};
    float m_run = -1e30f, l_run = 0.0f;

    stage(0, 0);

    for (int m0 = 0; m0 < MM_; m0 += 32) {
        int cur = (m0 >> 5) & 1;
        if (m0 + 32 < MM_) {
            stage(cur ^ 1, m0 + 32);
            wait_async<2>();   // current stage's 2 async ops (per wave) done
        } else {
            wait_async<0>();
        }
        __syncthreads();       // all waves' stage-cur LDS writes visible

        const _Float16* Kt = lds + cur * 4096;
        const _Float16* Vt = Kt + 2048;

        // --- S^T tiles from LDS: rows m, cols n ---
        const _Float16* k0 = Kt + (size_t)li * DD;
        const _Float16* k1 = Kt + (size_t)(16 + li) * DD;
        v16h a00 = mk16(*(const h8*)(k0 + hi * 8), *(const h8*)(k0 + 16 + hi * 8));
        v16h a01 = mk16(*(const h8*)(k0 + 32 + hi * 8), *(const h8*)(k0 + 48 + hi * 8));
        v16h a10 = mk16(*(const h8*)(k1 + hi * 8), *(const h8*)(k1 + 16 + hi * 8));
        v16h a11 = mk16(*(const h8*)(k1 + 32 + hi * 8), *(const h8*)(k1 + 48 + hi * 8));

        v8f s0 = {}, s1 = {};
        s0 = wmma_f16(a00, bq0, s0);
        s0 = wmma_f16(a01, bq1, s0);
        s1 = wmma_f16(a10, bq0, s1);
        s1 = wmma_f16(a11, bq1, s1);

        // --- mask + scale (32-bit byte tests); m = m0 + f*16 + hi*8 + r ---
        uint2 mw0 = *(const uint2*)(mbase + m0 + hi * 8);
        uint2 mw1 = *(const uint2*)(mbase + m0 + 16 + hi * 8);
        float sv0[8], sv1[8];
#pragma unroll
        for (int r = 0; r < 4; ++r) {
            sv0[r]     = msel(mw0.x, r, s0[r] * SCALE_F);
            sv0[r + 4] = msel(mw0.y, r, s0[r + 4] * SCALE_F);
            sv1[r]     = msel(mw1.x, r, s1[r] * SCALE_F);
            sv1[r + 4] = msel(mw1.y, r, s1[r + 4] * SCALE_F);
        }

        // --- online softmax over m: in-lane (8+8) + partner lane (xor 16) ---
        float tm = -1e30f;
#pragma unroll
        for (int r = 0; r < 8; ++r) tm = fmaxf(tm, fmaxf(sv0[r], sv1[r]));
        tm = fmaxf(tm, __shfl_xor(tm, 16, 32));
        float newm = fmaxf(m_run, tm);
        float corr = __expf(m_run - newm);

        float ts = 0.0f;
        H8U u0, u1;
#pragma unroll
        for (int r = 0; r < 8; ++r) {
            float p0 = __expf(sv0[r] - newm);
            float p1 = __expf(sv1[r] - newm);
            ts += p0 + p1;
            u0.h[r] = (_Float16)p0;
            u1.h[r] = (_Float16)p1;
        }
        ts += __shfl_xor(ts, 16, 32);
        l_run = l_run * corr + ts;
        m_run = newm;

        acc0 *= corr; acc1 *= corr; acc2 *= corr; acc3 *= corr;

        // --- assemble P^T B-frag (K = m, N = n) from the lane's own P (x0/x1 shuffles unused) ---
        H8U x0, x1;
#pragma unroll
        for (int wdd = 0; wdd < 4; ++wdd) {
            x0.u[wdd] = __shfl_xor(u0.u[wdd], 16, 32);
            x1.u[wdd] = __shfl_xor(u1.u[wdd], 16, 32);
        }
        h8 loH = u0.h;
        h8 hiH = u1.h;
        v16h bp = mk16(loH, hiH);

        // --- O^T += V^T * P^T : one wmma per 16-wide d tile (V from LDS) ---
        const _Float16* v0 = Vt + (size_t)(0 * 16 + li) * 32;
        const _Float16* v1 = Vt + (size_t)(1 * 16 + li) * 32;
        const _Float16* v2 = Vt + (size_t)(2 * 16 + li) * 32;
        const _Float16* v3 = Vt + (size_t)(3 * 16 + li) * 32;
        acc0 = wmma_f16(mk16(*(const h8*)(v0 + hi * 8), *(const h8*)(v0 + 16 + hi * 8)), bp, acc0);
        acc1 = wmma_f16(mk16(*(const h8*)(v1 + hi * 8), *(const h8*)(v1 + 16 + hi * 8)), bp, acc1);
        acc2 = wmma_f16(mk16(*(const h8*)(v2 + hi * 8), *(const h8*)(v2 + 16 + hi * 8)), bp, acc2);
        acc3 = wmma_f16(mk16(*(const h8*)(v3 + hi * 8), *(const h8*)(v3 + 16 + hi * 8)), bp, acc3);

        __syncthreads();       // everyone done reading buf `cur` before reuse
    }

    float inv = 1.0f / l_run;
    _Float16* obase = attnh + ((size_t)(b * NN_ + n)) * DIM_ + h * DD + hi * 8;
    v8f accs[4] = {acc0, acc1, acc2, acc3};
#pragma unroll
    for (int dt = 0; dt < 4; ++dt) {
        H8U o;
#pragma unroll
        for (int r = 0; r < 8; ++r) o.h[r] = (_Float16)(accs[dt][r] * inv);
        *(h8*)(obase + dt * 16) = o.h;
    }
}

// ---------------------------------------------------------------------------
// Host-side launch
// ---------------------------------------------------------------------------
extern "C" void kernel_launch(void* const* d_in, const int* in_sizes, int n_in,
                              void* d_out, int out_size, void* d_ws, size_t ws_size,
                              hipStream_t stream) {
    const float* x       = (const float*)d_in[0];
    const float* context = (const float*)d_in[1];
    const unsigned char* mask = (const unsigned char*)d_in[2];  // bool array
    const float* Wq  = (const float*)d_in[3];
    const float* Wkv = (const float*)d_in[4];
    const float* Wo  = (const float*)d_in[5];
    const float* qn_w = (const float*)d_in[6];
    const float* kn_w = (const float*)d_in[7];
    float* out = (float*)d_out;

    char* ws = (char*)d_ws;
    size_t off = 0;
    auto carve = [&](size_t bytes) {
        void* p = ws + off;
        off += (bytes + 255) & ~(size_t)255;
        return p;
    };

    _Float16* xh    = (_Float16*)carve((size_t)BB * NN_ * DIM_ * 2);        // 4 MB
    _Float16* ch    = (_Float16*)carve((size_t)BB * MM_ * DIM_ * 2);        // 8 MB
    _Float16* WqT   = (_Float16*)carve((size_t)DIM_ * DIM_ * 2);            // 2 MB
    _Float16* WkvT  = (_Float16*)carve((size_t)2 * DIM_ * DIM_ * 2);        // 4 MB
    _Float16* WoT   = (_Float16*)carve((size_t)DIM_ * DIM_ * 2);            // 2 MB
    _Float16* qraw  = (_Float16*)carve((size_t)BB * NN_ * DIM_ * 2);        // 4 MB
    _Float16* kvraw = (_Float16*)carve((size_t)BB * MM_ * 2 * DIM_ * 2);    // 16 MB
    _Float16* qh    = (_Float16*)carve((size_t)BB * HH * NN_ * DD * 2);     // 4 MB
    _Float16* kh    = (_Float16*)carve((size_t)BB * HH * MM_ * DD * 2);     // 8 MB
    _Float16* vT    = (_Float16*)carve((size_t)BB * HH * DD * MM_ * 2);     // 8 MB
    _Float16* attnh = (_Float16*)carve((size_t)BB * NN_ * DIM_ * 2);        // 4 MB
    (void)ws_size; (void)n_in; (void)in_sizes; (void)out_size;

    const int T = 256;  // 8 waves per block

    // 1) converts / weight transposes
    cvt_f32_to_f16<<<(BB * NN_ * DIM_ + T - 1) / T, T, 0, stream>>>(x, xh, BB * NN_ * DIM_);
    cvt_f32_to_f16<<<(BB * MM_ * DIM_ + T - 1) / T, T, 0, stream>>>(context, ch, BB * MM_ * DIM_);
    transpose_cvt<<<(DIM_ * DIM_ + T - 1) / T, T, 0, stream>>>(Wq, WqT, DIM_, DIM_);
    transpose_cvt<<<(DIM_ * 2 * DIM_ + T - 1) / T, T, 0, stream>>>(Wkv, WkvT, DIM_, 2 * DIM_);
    transpose_cvt<<<(DIM_ * DIM_ + T - 1) / T, T, 0, stream>>>(Wo, WoT, DIM_, DIM_);

    // 2) projection GEMMs (f16 out), block tile 128x64, async-LDS staged
    {   // q = x @ Wq : [2048 x 1024] x [1024 x 1024]
        int blocks = (BB * NN_ / 128) * (DIM_ / 64);     // 256
        gemm_wmma_lds<_Float16><<<blocks, T, 0, stream>>>(xh, WqT, qraw,
                                                          BB * NN_, DIM_, DIM_);
    }
    {   // kv = context @ Wkv : [4096 x 1024] x [1024 x 2048]
        int blocks = (BB * MM_ / 128) * (2 * DIM_ / 64); // 1024
        gemm_wmma_lds<_Float16><<<blocks, T, 0, stream>>>(ch, WkvT, kvraw,
                                                          BB * MM_, 2 * DIM_, DIM_);
    }

    // 3) per-head LN + V transpose
    ln_head<<<(BB * HH * NN_) / 8, T, 0, stream>>>(qraw, NN_, DIM_, qn_w, qh);
    ln_head<<<(BB * HH * MM_) / 8, T, 0, stream>>>(kvraw, MM_, 2 * DIM_, kn_w, kh);
    v_transpose<<<(BB * HH * DD * MM_ + T - 1) / T, T, 0, stream>>>(kvraw, vT);

    // 4) flash attention: 256 blocks x 256 threads, async-LDS staged K/V
    flash_attn<<<BB * HH * (NN_ / 128), T, 0, stream>>>(qh, kh, vT, mask, attnh);

    // 5) output projection (fp32 out)
    {
        int blocks = (BB * NN_ / 128) * (DIM_ / 64);     // 256
        gemm_wmma_lds<float><<<blocks, T, 0, stream>>>(attnh, WoT, out,
                                                       BB * NN_, DIM_, DIM_);
    }
}
